// DeformConv_1726576855893
// MI455X (gfx1250) — compile-verified, matrix operand order checked by reading, not run on hardware
//
#include <hip/hip_runtime.h>

typedef __bf16 bf16x16 __attribute__((ext_vector_type(16)));
typedef float  f32x8   __attribute__((ext_vector_type(8)));

#define CIN_     256
#define COUT_    256
#define HH_      64
#define WW_      64
#define HOWO_    4096
#define KTOT_    2304      // CIN*9
#define TILE_N_  64
#define C_CHUNK_ 32        // channels per K-chunk
#define K_CHUNK_ 288       // C_CHUNK*9
#define K_STRIDE_ 296      // padded LDS cols row stride (bf16 elems; 592B: 16B-aligned, conflict-free)

union FragBF {
    bf16x16 v;
    uint4   u[2];
};

__device__ __forceinline__ unsigned short f32_to_bf16_rne(float f) {
    unsigned int u = __float_as_uint(f);
    u += 0x7fffu + ((u >> 16) & 1u);
    return (unsigned short)(u >> 16);
}

// ---- prologue: weight fp32 -> bf16 (row-major [COUT, K], K = c*9 + kh*3 + kw) ----
__global__ void cvt_weight_bf16(const float* __restrict__ w,
                                unsigned short* __restrict__ o, int n) {
    int i = blockIdx.x * blockDim.x + threadIdx.x;
    if (i < n) o[i] = f32_to_bf16_rne(w[i]);
}

// ---- fused deformable-conv: bilinear im2col tile into LDS + WMMA GEMM ----
__launch_bounds__(256, 1)
__global__ void deform_conv_wmma(const float* __restrict__ x,
                                 const float* __restrict__ offs,
                                 const unsigned short* __restrict__ wbf,
                                 float* __restrict__ out) {
    __shared__ int4           s_idx[9 * TILE_N_];             // 4 clamped gather indices
    __shared__ float4         s_w[9 * TILE_N_];               // validity-folded bilinear weights
    __shared__ unsigned short s_cols[TILE_N_ * K_STRIDE_];    // bf16 cols tile [n][k]

    const int tid = threadIdx.x;
    const int b   = blockIdx.y;
    const int tn0 = blockIdx.x * TILE_N_;

    // ---------- phase 0: sample-point table (9 taps x 64 output positions) ----------
    for (int e = tid; e < 9 * TILE_N_; e += 256) {
        const int tap = e >> 6;
        const int n   = e & 63;
        const int ng  = tn0 + n;
        const int ho  = ng >> 6;
        const int wo  = ng & 63;
        const int kh  = tap / 3;
        const int kw  = tap - kh * 3;
        // offset layout: [B, KH, KW, 2, HO, WO]
        const float* op = offs + (size_t)(((b * 3 + kh) * 3 + kw) * 2) * HOWO_ + (ho << 6) + wo;
        const float offy = op[0];
        const float offx = op[HOWO_];

        const float py = (float)(kh + ho - 1) + offy;
        const float px = (float)(kw + wo - 1) + offx;
        const float y0f = floorf(py), x0f = floorf(px);
        const float ly = py - y0f,  lx = px - x0f;
        const float hy = 1.0f - ly, hx = 1.0f - lx;
        const int y0 = (int)y0f, x0 = (int)x0f;
        const int y1 = y0 + 1,   x1 = x0 + 1;
        const float vy0 = (y0 >= 0 && y0 < HH_) ? 1.0f : 0.0f;
        const float vy1 = (y1 >= 0 && y1 < HH_) ? 1.0f : 0.0f;
        const float vx0 = (x0 >= 0 && x0 < WW_) ? 1.0f : 0.0f;
        const float vx1 = (x1 >= 0 && x1 < WW_) ? 1.0f : 0.0f;
        const int cy0 = min(max(y0, 0), HH_ - 1);
        const int cy1 = min(max(y1, 0), HH_ - 1);
        const int cx0 = min(max(x0, 0), WW_ - 1);
        const int cx1 = min(max(x1, 0), WW_ - 1);
        s_idx[e] = make_int4(cy0 * WW_ + cx0, cy0 * WW_ + cx1,
                             cy1 * WW_ + cx0, cy1 * WW_ + cx1);
        s_w[e]   = make_float4(hy * hx * vy0 * vx0, hy * lx * vy0 * vx1,
                               ly * hx * vy1 * vx0, ly * lx * vy1 * vx1);
    }
    __syncthreads();

    // ---------- roles ----------
    const int lane   = tid & 31;
    const int l15    = lane & 15;
    const int khalf  = lane >> 4;          // K-half select per 16-bit WMMA layout
    const int wave   = tid >> 5;           // 8 waves
    const int m_base = wave * 2;           // 2 M blocks per wave (of 16)
                                           // every wave covers all 4 N blocks

    // per-wave A row base pointers (constant over the K loop)
    const unsigned short* arow[2];
    #pragma unroll
    for (int t = 0; t < 2; ++t)
        arow[t] = wbf + (size_t)((m_base + t) * 16 + l15) * KTOT_;

    // sampling role: 64 positions x 4 groups of 8 channels
    const int sn   = tid & 63;             // output position within tile
    const int c_l0 = (tid >> 6) * 8;       // first of 8 channels handled by this thread

    f32x8 acc[2][4] = {};

    for (int cc = 0; cc < CIN_ / C_CHUNK_; ++cc) {
        // ---------- phase 1: bilinear-sample cols chunk -> LDS (bf16) ----------
        #pragma unroll
        for (int cl = 0; cl < 8; ++cl) {
            const int c = cc * C_CHUNK_ + c_l0 + cl;
            const float* xb = x + ((size_t)(b * CIN_ + c) << 12);
            const int kbase = (c_l0 + cl) * 9;
            #pragma unroll
            for (int tap = 0; tap < 9; ++tap) {
                const int4   ii = s_idx[tap * TILE_N_ + sn];
                const float4 ww = s_w[tap * TILE_N_ + sn];
                const float v = ww.x * xb[ii.x] + ww.y * xb[ii.y]
                              + ww.z * xb[ii.z] + ww.w * xb[ii.w];
                s_cols[sn * K_STRIDE_ + kbase + tap] = f32_to_bf16_rne(v);
            }
        }
        __syncthreads();

        // ---------- phase 2: WMMA over this K chunk (9 k-blocks of 32) ----------
        const int Kg0 = cc * K_CHUNK_;
        #pragma unroll
        for (int kb = 0; kb < 9; ++kb) {
            const int koff = kb * 32 + khalf * 8;

            // B fragments (32x16 bf16) from LDS: all 4 N blocks
            FragBF bfr[4];
            #pragma unroll
            for (int nb = 0; nb < 4; ++nb) {
                const unsigned short* bp =
                    &s_cols[(nb * 16 + l15) * K_STRIDE_ + koff];
                bfr[nb].u[0] = *(const uint4*)(bp);
                bfr[nb].u[1] = *(const uint4*)(bp + 16);
            }

            // A fragments (16x32 bf16) from L2-resident bf16 weight: 2 M blocks
            const int Kg = Kg0 + koff;
            FragBF afr[2];
            #pragma unroll
            for (int t = 0; t < 2; ++t) {
                const unsigned short* ap = arow[t] + Kg;
                afr[t].u[0] = *(const uint4*)(ap);
                afr[t].u[1] = *(const uint4*)(ap + 16);
            }

            #pragma unroll
            for (int t = 0; t < 2; ++t)
                #pragma unroll
                for (int nb = 0; nb < 4; ++nb)
                    acc[t][nb] = __builtin_amdgcn_wmma_f32_16x16x32_bf16(
                        false, afr[t].v, false, bfr[nb].v,
                        (short)0, acc[t][nb], false, false);
        }
        __syncthreads();
    }

    // ---------- epilogue: C/D layout -> out[B, COUT, HO, WO] ----------
    #pragma unroll
    for (int t = 0; t < 2; ++t) {
        const int mb = m_base + t;
        #pragma unroll
        for (int nb = 0; nb < 4; ++nb) {
            const int ng = tn0 + nb * 16 + l15;
            #pragma unroll
            for (int r = 0; r < 8; ++r) {
                const int mg = mb * 16 + khalf * 8 + r;
                out[((size_t)(b * COUT_ + mg) << 12) + ng] = acc[t][nb][r];
            }
        }
    }
}

extern "C" void kernel_launch(void* const* d_in, const int* in_sizes, int n_in,
                              void* d_out, int out_size, void* d_ws, size_t ws_size,
                              hipStream_t stream) {
    (void)in_sizes; (void)n_in; (void)out_size; (void)ws_size;
    const float* x    = (const float*)d_in[0];
    const float* offs = (const float*)d_in[1];
    const float* w    = (const float*)d_in[2];
    unsigned short* wbf = (unsigned short*)d_ws;   // 256*2304*2 = 1.18 MB scratch

    const int nw = COUT_ * KTOT_;
    cvt_weight_bf16<<<(nw + 255) / 256, 256, 0, stream>>>(w, wbf, nw);

    dim3 grid(HOWO_ / TILE_N_, 4);                 // 64 n-tiles x 4 batches
    deform_conv_wmma<<<grid, 256, 0, stream>>>(x, offs, wbf, (float*)d_out);
}
